// LSTM_63797444214880
// MI455X (gfx1250) — compile-verified, matrix operand order checked by reading, not run on hardware
//
#include <hip/hip_runtime.h>
#include <hip/hip_bf16.h>

// ---------------------------------------------------------------------------
// LSTM on gfx1250 (MI455X): per-timestep fused GEMM+gate kernel.
//   u = [h | x_t] @ [W_hh | W_xh]^T + (b_hh + b_xh)       (K = 384)
// - v_wmma_f32_16x16x32_bf16, two accumulator chains (6 deep each)
// - h tile staged via GLOBAL_LOAD_ASYNC_TO_LDS_B128 (ASYNCcnt) once per
//   block, shared by all 4 gate-waves through ds_load_b128
// - W streams from L2 (768 KB bf16, resident in 192 MB L2 across all steps)
// - x / h_seq use non-temporal vmem ops (touched once, 128 MB / 256 MB)
// Shapes: T=2048, B=128, I=128, H=256, G=4H=1024.
// ---------------------------------------------------------------------------

typedef __attribute__((ext_vector_type(16))) __bf16 v16bf;
typedef __attribute__((ext_vector_type(8)))  __bf16 v8bf;
typedef __attribute__((ext_vector_type(8)))  float  v8f;
typedef __attribute__((ext_vector_type(4)))  float  v4f;

#define LSTM_T  2048
#define LSTM_B  128
#define LSTM_I  128
#define LSTM_H  256
#define LSTM_K  384   // H + I
#define LSTM_G  1024  // 4*H

#define ASH_STRIDE 264   // 256 + 8 bf16 pad; 528 B row = 16B multiple

// workspace layout (bytes)
#define WS_WC    0          // bf16 [1024][384] combined weights   (786432 B)
#define WS_BIAS  786432     // f32  [1024] combined bias           (  4096 B)
#define WS_HBF0  790528     // bf16 [128][256] h state, buffer 0   ( 65536 B)
#define WS_HBF1  856064     // bf16 [128][256] h state, buffer 1   ( 65536 B)
#define WS_CST   921600     // f32  [128][256] c state             (131072 B)

__device__ __forceinline__ float sigm_f(float x) {
    return 1.0f / (1.0f + __expf(-x));
}
__device__ __forceinline__ float tanh_f(float x) {
    float e = __expf(-2.0f * fabsf(x));
    float t = (1.0f - e) / (1.0f + e);
    return copysignf(t, x);
}

// ---------------------------------------------------------------------------
// Prep: bf16-pack combined weights [W_hh | W_xh], combine biases, init state.
// ---------------------------------------------------------------------------
__global__ void lstm_prep(const float* __restrict__ Whh,
                          const float* __restrict__ Wxh,
                          const float* __restrict__ bhh,
                          const float* __restrict__ bxh,
                          const float* __restrict__ h0,
                          const float* __restrict__ c0,
                          __bf16* __restrict__ Wc,
                          float*  __restrict__ bias,
                          __bf16* __restrict__ hbf0,
                          float*  __restrict__ cst)
{
    int idx = blockIdx.x * 256 + threadIdx.x;
    if (idx < LSTM_G * LSTM_K) {                       // 393216: weight pack
        int n = idx / LSTM_K, k = idx % LSTM_K;
        float v = (k < LSTM_H) ? Whh[n * LSTM_H + k]
                               : Wxh[n * LSTM_I + (k - LSTM_H)];
        Wc[idx] = (__bf16)v;
    } else if (idx < 393216 + LSTM_G) {                // bias combine
        int n = idx - 393216;
        bias[n] = bhh[n] + bxh[n];
    } else if (idx < 394240 + LSTM_B * LSTM_H) {       // h0 -> bf16
        int e = idx - 394240;
        hbf0[e] = (__bf16)h0[e];
    } else if (idx < 427008 + LSTM_B * LSTM_H) {       // c0 -> f32 state
        int e = idx - 427008;
        cst[e] = c0[e];
    }
}

// ---------------------------------------------------------------------------
// One timestep. grid = (16 H-tiles, 8 batch-tiles), block = 128 (4 waves).
// Wave g computes gate g's 16x16 u-tile with 12 bf16 WMMAs (K=384) in two
// 6-deep accumulator chains, then the block fuses the cell update via LDS.
// ---------------------------------------------------------------------------
__global__ __launch_bounds__(128) void lstm_step(
    const float*  __restrict__ xt,    // [128][128] f32, this timestep
    const __bf16* __restrict__ Wc,    // [1024][384] bf16
    const float*  __restrict__ bias,  // [1024]
    const __bf16* __restrict__ hin,   // [128][256] bf16 (prev h)
    __bf16*       __restrict__ hout_bf,// [128][256] bf16 (new h)
    float*        __restrict__ cst,   // [128][256] f32 (c state, in/out)
    float*        __restrict__ hout)  // [128][256] f32 -> h_seq[t]
{
    __shared__ __bf16 Ash[16][ASH_STRIDE];  // h tile (recurrent half of A)
    __shared__ float  Us[4][16][16];        // per-gate u tiles for exchange

    const int jt   = blockIdx.x;          // H tile: 0..15
    const int mt   = blockIdx.y;          // batch tile: 0..7
    const int tid  = threadIdx.x;
    const int lane = tid & 31;
    const int gate = tid >> 5;            // 0=f 1=i 2=o 3=g
    const int nl   = lane & 15;
    const int half = lane >> 4;           // K-half per ISA 16-bit A layout

    // ---- async-stage the 16x256 bf16 h tile into LDS (512 x 16B chunks) ---
    {
        const unsigned ldsA = (unsigned)(size_t)(&Ash[0][0]);
        #pragma unroll
        for (int q = 0; q < 4; ++q) {
            const int chunk = q * 128 + tid;      // 0..511
            const int row   = chunk >> 5;         // 0..15
            const int k     = (chunk & 31) * 8;   // 0..248
            const unsigned lds_dst = ldsA + (unsigned)(row * ASH_STRIDE + k) * 2u;
            const unsigned long long gaddr =
                (unsigned long long)(size_t)(hin + (size_t)(mt * 16 + row) * LSTM_H + k);
            asm volatile("global_load_async_to_lds_b128 %0, %1, off"
                         :: "v"(lds_dst), "v"(gaddr) : "memory");
        }
        asm volatile("s_wait_asynccnt 0x0" ::: "memory");
    }
    __syncthreads();

    const float*  xrow = xt + (size_t)(mt * 16 + nl) * LSTM_I;
    const __bf16* wr   = Wc + (size_t)(gate * LSTM_H + jt * 16 + nl) * LSTM_K;
    const __bf16* arow = &Ash[nl][0];

    __builtin_prefetch(wr, 0, 0);        // global_prefetch_b8

    v8f acc0 = {}, acc1 = {};
    #pragma unroll
    for (int kk = 0; kk < LSTM_K; kk += 32) {
        const int k0 = kk + half * 8;         // lane's first 8-elem chunk
        const int k1 = kk + 16 + half * 8;    // lane's second 8-elem chunk

        v8bf a0, a1;
        if (kk < LSTM_H) {
            // recurrent half of A: bf16 h tile from LDS (ds_load_b128)
            a0 = *(const v8bf*)(arow + k0);
            a1 = *(const v8bf*)(arow + k1);
        } else {
            // input half of A: f32 x streamed non-temporally, cvt to bf16
            const v4f* p0 = (const v4f*)(xrow + (k0 - LSTM_H));
            const v4f* p1 = (const v4f*)(xrow + (k1 - LSTM_H));
            v4f f0 = __builtin_nontemporal_load(p0);
            v4f f1 = __builtin_nontemporal_load(p0 + 1);
            v4f f2 = __builtin_nontemporal_load(p1);
            v4f f3 = __builtin_nontemporal_load(p1 + 1);
            a0 = (v8bf){(__bf16)f0.x, (__bf16)f0.y, (__bf16)f0.z, (__bf16)f0.w,
                        (__bf16)f1.x, (__bf16)f1.y, (__bf16)f1.z, (__bf16)f1.w};
            a1 = (v8bf){(__bf16)f2.x, (__bf16)f2.y, (__bf16)f2.z, (__bf16)f2.w,
                        (__bf16)f3.x, (__bf16)f3.y, (__bf16)f3.z, (__bf16)f3.w};
        }
        v8bf b0 = *(const v8bf*)(wr + k0);
        v8bf b1 = *(const v8bf*)(wr + k1);

        v16bf av, bv;
        #pragma unroll
        for (int i = 0; i < 8; ++i) {
            av[i] = a0[i]; av[i + 8] = a1[i];
            bv[i] = b0[i]; bv[i + 8] = b1[i];
        }
        // D = A(16x32 bf16) x B(32x16 bf16) + C(16x16 f32); two chains
        if (((kk >> 5) & 1) == 0)
            acc0 = __builtin_amdgcn_wmma_f32_16x16x32_bf16(
                false, av, false, bv, (short)0, acc0, false, false);
        else
            acc1 = __builtin_amdgcn_wmma_f32_16x16x32_bf16(
                false, av, false, bv, (short)0, acc1, false, false);
    }
    const v8f acc = acc0 + acc1;

    // bias is constant per output column (== per lane here)
    const float bb = bias[gate * LSTM_H + jt * 16 + nl];
    // C/D layout: VGPR v -> M = v + 8*half, N = lane&15
    #pragma unroll
    for (int v = 0; v < 8; ++v)
        Us[gate][v + half * 8][nl] = acc[v] + bb;

    __syncthreads();

    // fused cell update: 256 elements per (mt,jt) tile, 2 per thread
    #pragma unroll
    for (int e = tid; e < 256; e += 128) {
        const int r  = e >> 4, cj = e & 15;
        const int b  = mt * 16 + r;
        const int j  = jt * 16 + cj;
        const float f = sigm_f(Us[0][r][cj]);
        const float i = sigm_f(Us[1][r][cj]);
        const float o = sigm_f(Us[2][r][cj]);
        const float g = tanh_f(Us[3][r][cj]);
        const int   idx = b * LSTM_H + j;
        const float cnew = cst[idx] * f + g * i;
        const float hnew = o * tanh_f(cnew);
        cst[idx]     = cnew;
        hout_bf[idx] = (__bf16)hnew;
        __builtin_nontemporal_store(hnew, &hout[idx]);   // h_seq streamed out
    }
}

// ---------------------------------------------------------------------------
// Final: hT = h_seq[T-1]; cT = c state.   d_out = [h_seq | hT | cT].
// ---------------------------------------------------------------------------
__global__ void lstm_final(float* __restrict__ out, const float* __restrict__ cst)
{
    const size_t HS = (size_t)LSTM_T * LSTM_B * LSTM_H;   // 67108864
    int idx = blockIdx.x * 256 + threadIdx.x;             // 0..65535
    if (idx < LSTM_B * LSTM_H)
        out[HS + idx] = out[HS - LSTM_B * LSTM_H + idx];
    else
        out[HS + idx] = cst[idx - LSTM_B * LSTM_H];
}

extern "C" void kernel_launch(void* const* d_in, const int* in_sizes, int n_in,
                              void* d_out, int out_size, void* d_ws, size_t ws_size,
                              hipStream_t stream) {
    const float* x    = (const float*)d_in[0];
    const float* h0   = (const float*)d_in[1];
    const float* c0   = (const float*)d_in[2];
    const float* W_hh = (const float*)d_in[3];
    const float* b_hh = (const float*)d_in[4];
    const float* W_xh = (const float*)d_in[5];
    const float* b_xh = (const float*)d_in[6];
    float* out = (float*)d_out;

    char*   ws   = (char*)d_ws;
    __bf16* Wc   = (__bf16*)(ws + WS_WC);
    float*  bias = (float*)(ws + WS_BIAS);
    __bf16* hbf0 = (__bf16*)(ws + WS_HBF0);
    __bf16* hbf1 = (__bf16*)(ws + WS_HBF1);
    float*  cst  = (float*)(ws + WS_CST);

    lstm_prep<<<dim3(1796), dim3(256), 0, stream>>>(
        W_hh, W_xh, b_hh, b_xh, h0, c0, Wc, bias, hbf0, cst);

    for (int t = 0; t < LSTM_T; ++t) {
        __bf16* hin  = (t & 1) ? hbf1 : hbf0;
        __bf16* hnew = (t & 1) ? hbf0 : hbf1;
        lstm_step<<<dim3(16, 8), dim3(128), 0, stream>>>(
            x + (size_t)t * LSTM_B * LSTM_I, Wc, bias,
            hin, hnew, cst,
            out + (size_t)t * LSTM_B * LSTM_H);
    }

    lstm_final<<<dim3(256), dim3(256), 0, stream>>>(out, cst);
}
